// MultiHeadAttention_45286135169102
// MI455X (gfx1250) — matrix operand order repaired and checked by reading, not compiled, not run on hardware
//
#include <hip/hip_runtime.h>
#include <hip/hip_bf16.h>

// ---------------- problem constants ----------------
#define B_   2
#define S_   2048
#define E_   1024
#define H_   16
#define DH_  64
#define M_   (B_ * S_)          // 4096 rows
static constexpr float SCALE_ = 0.125f;   // DH^-0.5

// ---------------- vector types ----------------
typedef __attribute__((ext_vector_type(16))) __bf16 v16bf;
typedef __attribute__((ext_vector_type(8)))  __bf16 v8bf;
typedef __attribute__((ext_vector_type(4)))  __bf16 v4bf;
typedef __attribute__((ext_vector_type(8)))  float  v8f;
typedef __attribute__((vector_size(16)))     int    v4i_;

#if __has_builtin(__builtin_amdgcn_global_load_async_to_lds_b128)
#define HAVE_ASYNC_LDS 1
#endif
#if __has_builtin(__builtin_amdgcn_update_dpp)
#define HAVE_DPP 1
#endif

__device__ __forceinline__ __bf16 f2bf(float f) {
  unsigned u = __builtin_bit_cast(unsigned, f);
  unsigned r = u + 0x7fffu + ((u >> 16) & 1u);     // round-to-nearest-even
  unsigned short h = (unsigned short)(r >> 16);
  return __builtin_bit_cast(__bf16, h);
}

__device__ __forceinline__ v8f wmma_bf16(v16bf a, v16bf b, v8f c) {
  return __builtin_amdgcn_wmma_f32_16x16x32_bf16(
      /*neg_a=*/false, a, /*neg_b=*/false, b,
      /*c_mod=*/(short)0, c, /*reuse_a=*/false, /*reuse_b=*/false);
}

__device__ __forceinline__ void wait_async0() {
#ifdef HAVE_ASYNC_LDS
#if __has_builtin(__builtin_amdgcn_s_wait_asynccnt)
  __builtin_amdgcn_s_wait_asynccnt(0);
#else
  asm volatile("s_wait_asynccnt 0" ::: "memory");
#endif
#endif
}

// Max-reduce across each 16-lane half of the wave (rows never cross halves).
__device__ __forceinline__ float redmax16(float x) {
#ifdef HAVE_DPP
  // quad_perm xor1 (0xB1), quad_perm xor2 (0x4E), row_half_mirror (0x141),
  // row_mirror (0x140): VALU-only butterfly, stays within 16-lane DPP rows.
  int xi = __builtin_bit_cast(int, x);
  float t;
  t = __builtin_bit_cast(float, __builtin_amdgcn_update_dpp(0, xi, 0xB1, 0xF, 0xF, true));
  x = fmaxf(x, t); xi = __builtin_bit_cast(int, x);
  t = __builtin_bit_cast(float, __builtin_amdgcn_update_dpp(0, xi, 0x4E, 0xF, 0xF, true));
  x = fmaxf(x, t); xi = __builtin_bit_cast(int, x);
  t = __builtin_bit_cast(float, __builtin_amdgcn_update_dpp(0, xi, 0x141, 0xF, 0xF, true));
  x = fmaxf(x, t); xi = __builtin_bit_cast(int, x);
  t = __builtin_bit_cast(float, __builtin_amdgcn_update_dpp(0, xi, 0x140, 0xF, 0xF, true));
  x = fmaxf(x, t);
  return x;
#else
#pragma unroll
  for (int d = 8; d >= 1; d >>= 1) x = fmaxf(x, __shfl_xor(x, d, 32));
  return x;
#endif
}

// ---------------- fp32 -> bf16 cast (x4 vectorized) ----------------
__global__ void cast_f32_bf16_x4(const float* __restrict__ s,
                                 __bf16* __restrict__ d, int n4) {
  int i = blockIdx.x * blockDim.x + threadIdx.x;
  if (i >= n4) return;
  const float4 v = reinterpret_cast<const float4*>(s)[i];
  v4bf o;
  o[0] = f2bf(v.x); o[1] = f2bf(v.y); o[2] = f2bf(v.z); o[3] = f2bf(v.w);
  reinterpret_cast<v4bf*>(d)[i] = o;
}

// ---------------- GEMM: C = (A @ W^T + bias) * alpha ----------------
// A: [Mdim x Kdim] bf16 row-major, W: [Ndim x Kdim] bf16 row-major.
// Block: 256 threads = 8 waves; wave tile = 16(M) x 64(N); block = 128 x 64.
template<bool OUT_BF16>
__global__ __launch_bounds__(256)
void gemm_bias(const __bf16* __restrict__ A, const __bf16* __restrict__ W,
               const float* __restrict__ bias, float alpha,
               void* __restrict__ Cout, int Mdim, int Ndim, int Kdim) {
  const int lane  = threadIdx.x & 31;
  const int wv    = threadIdx.x >> 5;
  const int m0    = blockIdx.x * 128 + wv * 16;
  const int n0    = blockIdx.y * 64;
  const int nlane = lane & 15;
  const int kb    = ((lane >> 4) & 1) * 8;   // unused; A-frag chunks are based at khalf
  const int khalf = (lane >> 4) * 16;        // B-frag K base within K32

  v8f c0 = {}, c1 = {}, c2 = {}, c3 = {};

  const __bf16* arow = A + (size_t)(m0 + nlane) * Kdim;
  const __bf16* w0 = W + (size_t)(n0      + nlane) * Kdim + khalf;
  const __bf16* w1 = W + (size_t)(n0 + 16 + nlane) * Kdim + khalf;
  const __bf16* w2 = W + (size_t)(n0 + 32 + nlane) * Kdim + khalf;
  const __bf16* w3 = W + (size_t)(n0 + 48 + nlane) * Kdim + khalf;

  for (int k0 = 0; k0 < Kdim; k0 += 32) {
    __builtin_prefetch(arow + k0 + 256, 0, 0);
    __builtin_prefetch(w0 + k0 + 256, 0, 0);
    // A fragment: lane half selects K offset 0/16,
    // element chunks cover [khalf, khalf+8) and [khalf+8, khalf+16).
    v8bf alo = *(const v8bf*)(arow + k0 + khalf);
    v8bf ahi = *(const v8bf*)(arow + k0 + khalf + 8);
    v16bf a;
#pragma unroll
    for (int i = 0; i < 8; ++i) { a[i] = alo[i]; a[8 + i] = ahi[i]; }
    // B fragments: lane holds column n = n0+t*16+nlane; 16 contiguous K values.
    v16bf b0 = *(const v16bf*)(w0 + k0);
    v16bf b1 = *(const v16bf*)(w1 + k0);
    v16bf b2 = *(const v16bf*)(w2 + k0);
    v16bf b3 = *(const v16bf*)(w3 + k0);
    c0 = wmma_bf16(a, b0, c0);
    c1 = wmma_bf16(a, b1, c1);
    c2 = wmma_bf16(a, b2, c2);
    c3 = wmma_bf16(a, b3, c3);
  }

  // Epilogue: C layout row m = r + 8*(lane>=16), col n = nlane.
  const int rbase = (lane >> 4) * 8;
#pragma unroll
  for (int t = 0; t < 4; ++t) {
    v8f c = (t == 0) ? c0 : (t == 1) ? c1 : (t == 2) ? c2 : c3;
    const int n = n0 + t * 16 + nlane;
    const float bb = bias[n];
#pragma unroll
    for (int r = 0; r < 8; ++r) {
      float val = (c[r] + bb) * alpha;
      size_t idx = (size_t)(m0 + rbase + r) * Ndim + n;
      if (OUT_BF16) ((__bf16*)Cout)[idx] = f2bf(val);
      else          ((float*)Cout)[idx]  = val;
    }
  }
}

// ---------------- fused flash attention (double-buffered) ----------------
// Grid: (S/128, B*H). Block: 8 waves sharing one (b,h); wave w owns queries
// [q0, q0+16). 32-key tiles, online softmax. K tile staged by async DMA
// (ASYNCcnt), double-buffered so tile j+1 loads overlap compute on tile j.
// Row-sums of P come from an extra WMMA against a ones matrix.
__global__ __launch_bounds__(256)
void attention_kernel(const __bf16* __restrict__ Q, const __bf16* __restrict__ K,
                      const __bf16* __restrict__ V, __bf16* __restrict__ O) {
  __shared__ __bf16 Kt[2][32][DH_];    // key tiles                  (8 KB)
  __shared__ __bf16 Vtt[2][DH_][32];   // transposed value tiles     (8 KB)
  __shared__ float  Ps[8][16][32];     // per-wave P re-swizzle     (16 KB)

  const int lane  = threadIdx.x & 31;
  const int wv    = threadIdx.x >> 5;
  const int b     = blockIdx.y / H_;
  const int h     = blockIdx.y % H_;
  const int q0    = blockIdx.x * 128 + wv * 16;
  const size_t rowbase = (size_t)b * S_;

  const int nlane = lane & 15;
  const int kb    = ((lane >> 4) & 1) * 8;
  const int khalf = (lane >> 4) * 16;
  const int rbase = (lane >> 4) * 8;

  // Preload Q A-fragments for the two 32-wide K-chunks of DH=64.
  const __bf16* qrow = Q + (rowbase + q0 + nlane) * E_ + h * DH_;
  v16bf aq[2];
#pragma unroll
  for (int kc = 0; kc < 2; ++kc) {
    v8bf lo = *(const v8bf*)(qrow + kc * 32 + khalf);
    v8bf hi = *(const v8bf*)(qrow + kc * 32 + khalf + 8);
#pragma unroll
    for (int i = 0; i < 8; ++i) { aq[kc][i] = lo[i]; aq[kc][8 + i] = hi[i]; }
  }

  // All-ones B fragment for row-sum WMMA.
  v16bf onesb;
  {
    const __bf16 one = f2bf(1.0f);
#pragma unroll
    for (int e = 0; e < 16; ++e) onesb[e] = one;
  }

  v8f o0 = {}, o1 = {}, o2 = {}, o3 = {};
  v8f lacc = {};                    // softmax denominators (row-sums)
  float ms[8];
#pragma unroll
  for (int r = 0; r < 8; ++r) ms[r] = -__builtin_inff();

  const int sr = threadIdx.x >> 3;        // staging key row 0..31
  const int sc = (threadIdx.x & 7) * 8;   // staging dim base

  // Prologue: stage tile 0 into buffer 0.
  {
    const size_t grow = (rowbase + sr) * E_ + h * DH_ + sc;
#ifdef HAVE_ASYNC_LDS
    __builtin_amdgcn_global_load_async_to_lds_b128(
        (__attribute__((address_space(1))) v4i_*)(K + grow),
        (__attribute__((address_space(3))) v4i_*)&Kt[0][sr][sc], 0, 0);
#else
    *(v8bf*)&Kt[0][sr][sc] = *(const v8bf*)(K + grow);
#endif
    v8bf vvv = *(const v8bf*)(V + grow);
#pragma unroll
    for (int i = 0; i < 8; ++i) Vtt[0][sc + i][sr] = vvv[i];
    wait_async0();
  }
  __syncthreads();

  constexpr int NJ = S_ / 32;
  for (int j = 0; j < NJ; ++j) {
    const int cur = j & 1;
    const int nxt = cur ^ 1;
    const bool have_next = (j + 1 < NJ);

    // Kick off tile j+1: async K DMA + V global load (overlaps compute).
    // Buffer `nxt` was last READ in iteration j-1; the barrier at the end of
    // j-1 ordered those reads before these writes.
    v8bf vnext;
    if (have_next) {
      const size_t grow = (rowbase + (j + 1) * 32 + sr) * E_ + h * DH_ + sc;
#ifdef HAVE_ASYNC_LDS
      __builtin_amdgcn_global_load_async_to_lds_b128(
          (__attribute__((address_space(1))) v4i_*)(K + grow),
          (__attribute__((address_space(3))) v4i_*)&Kt[nxt][sr][sc], 0, 0);
#else
      *(v8bf*)&Kt[nxt][sr][sc] = *(const v8bf*)(K + grow);
#endif
      vnext = *(const v8bf*)(V + grow);
    }

    // Scores: S(16x32) = Q(16x64) @ K_tile^T. Two 16-wide N halves.
    v8f s0 = {}, s1 = {};
    {
      v16bf bk;
      bk = *(const v16bf*)&Kt[cur][nlane][khalf];           s0 = wmma_bf16(aq[0], bk, s0);
      bk = *(const v16bf*)&Kt[cur][nlane][32 + khalf];      s0 = wmma_bf16(aq[1], bk, s0);
      bk = *(const v16bf*)&Kt[cur][16 + nlane][khalf];      s1 = wmma_bf16(aq[0], bk, s1);
      bk = *(const v16bf*)&Kt[cur][16 + nlane][32 + khalf]; s1 = wmma_bf16(aq[1], bk, s1);
    }

    // Online softmax: DPP max-reduce per row (rows live in 16-lane halves).
#pragma unroll
    for (int r = 0; r < 8; ++r) {
      const float t = redmax16(fmaxf(s0[r], s1[r]));
      const float mnew = fmaxf(ms[r], t);
      const float corr = __expf(ms[r] - mnew);
      ms[r] = mnew;
      s0[r] = __expf(s0[r] - mnew);
      s1[r] = __expf(s1[r] - mnew);
      lacc[r] *= corr;
      o0[r] *= corr; o1[r] *= corr; o2[r] *= corr; o3[r] *= corr;
    }

    // Re-swizzle P: fp32 C-layout -> bf16 A-layout via per-wave LDS scratch.
    // No barrier needed: Ps is wave-private and same-wave LDS ops are in-order.
#pragma unroll
    for (int r = 0; r < 8; ++r) {
      Ps[wv][rbase + r][nlane]      = s0[r];
      Ps[wv][rbase + r][16 + nlane] = s1[r];
    }
    v16bf pa;
#pragma unroll
    for (int e = 0; e < 16; ++e) {
      const int vv2 = e >> 1, hh = e & 1;
      const int kk = khalf + vv2 * 2 + hh;
      pa[e] = f2bf(Ps[wv][nlane][kk]);
    }

    // Row-sum of P via ones-matrix WMMA (replaces shuffle sum reduction).
    lacc = wmma_bf16(pa, onesb, lacc);

    // O += P(16x32) @ V_tile(32x64): 4 wmma over the 4 d-subtiles.
    v16bf bvf;
    bvf = *(const v16bf*)&Vtt[cur][ 0 + nlane][khalf]; o0 = wmma_bf16(pa, bvf, o0);
    bvf = *(const v16bf*)&Vtt[cur][16 + nlane][khalf]; o1 = wmma_bf16(pa, bvf, o1);
    bvf = *(const v16bf*)&Vtt[cur][32 + nlane][khalf]; o2 = wmma_bf16(pa, bvf, o2);
    bvf = *(const v16bf*)&Vtt[cur][48 + nlane][khalf]; o3 = wmma_bf16(pa, bvf, o3);

    // Transpose-store the prefetched V tile into the other buffer.
    if (have_next) {
#pragma unroll
      for (int i = 0; i < 8; ++i) Vtt[nxt][sc + i][sr] = vnext[i];
    }

    wait_async0();       // tile j+1 K DMA complete (overlapped with compute)
    __syncthreads();     // all waves done with buffer `cur`; `nxt` now ready
  }

  // Normalize and write attention output (bf16, [B*S, E] with head offset).
#pragma unroll
  for (int t = 0; t < 4; ++t) {
    v8f oc = (t == 0) ? o0 : (t == 1) ? o1 : (t == 2) ? o2 : o3;
#pragma unroll
    for (int r = 0; r < 8; ++r) {
      const float val = oc[r] / lacc[r];
      O[(rowbase + q0 + rbase + r) * E_ + h * DH_ + t * 16 + nlane] = f2bf(val);
    }
  }
}

// ---------------- host launcher ----------------
extern "C" void kernel_launch(void* const* d_in, const int* in_sizes, int n_in,
                              void* d_out, int out_size, void* d_ws, size_t ws_size,
                              hipStream_t stream) {
  const float* x  = (const float*)d_in[0];
  const float* Wq = (const float*)d_in[1];
  const float* bq = (const float*)d_in[2];
  const float* Wk = (const float*)d_in[3];
  const float* bk = (const float*)d_in[4];
  const float* Wv = (const float*)d_in[5];
  const float* bv = (const float*)d_in[6];
  const float* Wo = (const float*)d_in[7];
  const float* bo = (const float*)d_in[8];

  char* ws = (char*)d_ws;
  size_t off = 0;
  auto alloc = [&](size_t bytes) -> char* {
    char* p = ws + off;
    off += (bytes + 255) & ~(size_t)255;
    return p;
  };
  __bf16* xb  = (__bf16*)alloc((size_t)M_ * E_ * 2);
  __bf16* Wqb = (__bf16*)alloc((size_t)E_ * E_ * 2);
  __bf16* Wkb = (__bf16*)alloc((size_t)E_ * E_ * 2);
  __bf16* Wvb = (__bf16*)alloc((size_t)E_ * E_ * 2);
  __bf16* Wob = (__bf16*)alloc((size_t)E_ * E_ * 2);
  __bf16* Qb  = (__bf16*)alloc((size_t)M_ * E_ * 2);
  __bf16* Kb  = (__bf16*)alloc((size_t)M_ * E_ * 2);
  __bf16* Vb  = (__bf16*)alloc((size_t)M_ * E_ * 2);
  __bf16* Ob  = (__bf16*)alloc((size_t)M_ * E_ * 2);
  (void)ws_size; (void)in_sizes; (void)n_in; (void)out_size;

  auto cast = [&](const float* s, __bf16* d, int n) {
    int n4 = n / 4;
    cast_f32_bf16_x4<<<dim3((n4 + 255) / 256), dim3(256), 0, stream>>>(s, d, n4);
  };
  cast(x,  xb,  M_ * E_);
  cast(Wq, Wqb, E_ * E_);
  cast(Wk, Wkb, E_ * E_);
  cast(Wv, Wvb, E_ * E_);
  cast(Wo, Wob, E_ * E_);

  dim3 g(M_ / 128, E_ / 64), blk(256);
  gemm_bias<true><<<g, blk, 0, stream>>>(xb, Wqb, bq, SCALE_, (void*)Qb, M_, E_, E_);
  gemm_bias<true><<<g, blk, 0, stream>>>(xb, Wkb, bk, 1.0f,   (void*)Kb, M_, E_, E_);
  gemm_bias<true><<<g, blk, 0, stream>>>(xb, Wvb, bv, 1.0f,   (void*)Vb, M_, E_, E_);

  attention_kernel<<<dim3(S_ / 128, B_ * H_), blk, 0, stream>>>(Qb, Kb, Vb, Ob);

  gemm_bias<false><<<g, blk, 0, stream>>>(Ob, Wob, bo, 1.0f, d_out, M_, E_, E_);
}
